// MockSparseModel_24532853195121
// MI455X (gfx1250) — compile-verified, matrix operand order checked by reading, not run on hardware
//
#include <hip/hip_runtime.h>
#include <hip/hip_bf16.h>

typedef __attribute__((ext_vector_type(16))) _Float16 v16h;
typedef __attribute__((ext_vector_type(8)))  float    v8f;
typedef __attribute__((ext_vector_type(4)))  float    vf4;

#define BLOCK 256

// Materialize the 8-VGPR f32 zero tile with one matrix op (0*0+0).
// EXEC is all 1s at call sites (no divergence has occurred yet).
__device__ __forceinline__ vf4 wmma_zero_f32x4() {
  v16h za = {};
  v8f  zc = {};
  zc = __builtin_amdgcn_wmma_f32_16x16x32_f16(
      /*neg_a=*/false, za, /*neg_b=*/false, za,
      /*c_mod=*/(short)0, zc, /*reuse_a=*/false, /*reuse_b=*/false);
  vf4 z;
  z[0] = zc[0]; z[1] = zc[1]; z[2] = zc[2]; z[3] = zc[3];
  return z;
}

// Compile-time-vocab specialization: trip count is a constant (VOCAB/4/BLOCK),
// fully unrolled -> one base address + immediate-offset b128 NT store clauses,
// no peel loop, no per-iteration address math.
template <int VOCAB>
__global__ __launch_bounds__(BLOCK) void onehot_boost_fixed(
    const int*   __restrict__ ids,
    const int*   __restrict__ msk,
    const float* __restrict__ boost,
    float*       __restrict__ out) {
  static_assert(VOCAB % (4 * BLOCK) == 0, "row must tile evenly");
  const int row = blockIdx.x;                 // uniform -> scalar loads
  const int id  = ids[row];
  const int m   = msk[row];
  const bool valid = (m == 1) && (id >= 0) && (id < VOCAB);

  const vf4 z0 = wmma_zero_f32x4();

  float* __restrict__ rowp  = out + (size_t)row * (size_t)VOCAB;
  vf4*   __restrict__ rowp4 = (vf4*)rowp;

  constexpr int ITERS = (VOCAB / 4) / BLOCK;  // 32 for VOCAB=32768
  #pragma unroll
  for (int k = 0; k < ITERS; ++k) {
    const int j = (int)threadIdx.x + k * BLOCK;   // constant-folded offsets
    __builtin_nontemporal_store(z0, rowp4 + j);
  }

  // The lane that zeroed the target float4 rewrites the scattered element;
  // same-wave stores to the same address stay in order (CDNA5 ISA), no sync.
  if (valid && (((unsigned)(id >> 2)) % BLOCK) == threadIdx.x) {
    rowp[id] = boost[0];
  }
}

// Generic runtime-vocab fallback (correct for any shape).
__global__ __launch_bounds__(BLOCK) void onehot_boost_generic(
    const int*   __restrict__ ids,
    const int*   __restrict__ msk,
    const float* __restrict__ boost,
    float*       __restrict__ out,
    int vocab) {
  const int row = blockIdx.x;
  const int id  = ids[row];
  const int m   = msk[row];
  const bool valid = (m == 1) && (id >= 0) && (id < vocab);

  const vf4 z0 = wmma_zero_f32x4();

  float* __restrict__ rowp  = out + (size_t)row * (size_t)vocab;
  vf4*   __restrict__ rowp4 = (vf4*)rowp;
  const int n4 = vocab >> 2;

  #pragma unroll 8
  for (int j = (int)threadIdx.x; j < n4; j += BLOCK) {
    __builtin_nontemporal_store(z0, rowp4 + j);
  }
  // Tail floats if vocab % 4 != 0 (not hit for VOCAB=32768).
  for (int j = (n4 << 2) + (int)threadIdx.x; j < vocab; j += BLOCK) {
    __builtin_nontemporal_store(0.0f, rowp + j);
  }

  if (valid && (((unsigned)(id >> 2)) % BLOCK) == threadIdx.x) {
    rowp[id] = boost[0];
  }
}

extern "C" void kernel_launch(void* const* d_in, const int* in_sizes, int n_in,
                              void* d_out, int out_size, void* d_ws, size_t ws_size,
                              hipStream_t stream) {
  const int*   ids   = (const int*)d_in[0];   // input_ids, int32 [B*S]
  const int*   msk   = (const int*)d_in[1];   // attention_mask, int32 [B*S]
  const float* boost = (const float*)d_in[2]; // boost, f32 [1]
  float*       out   = (float*)d_out;         // [B*S*V] f32

  const int rows  = in_sizes[0];              // B*S = 4096
  const int vocab = out_size / rows;          // 32768

  dim3 grid((unsigned)rows), block(BLOCK);
  if (vocab == 32768) {
    hipLaunchKernelGGL((onehot_boost_fixed<32768>), grid, block, 0, stream,
                       ids, msk, boost, out);
  } else {
    hipLaunchKernelGGL(onehot_boost_generic, grid, block, 0, stream,
                       ids, msk, boost, out, vocab);
  }
}
